// HindiCausalLMAttention_59863254171992
// MI455X (gfx1250) — matrix operand order repaired and checked by reading, not compiled, not run on hardware
//
#include <hip/hip_runtime.h>
#include <math.h>

typedef _Float16 f16;
typedef __attribute__((ext_vector_type(16))) _Float16 v16h;
typedef __attribute__((ext_vector_type(8)))  _Float16 h8;
typedef __attribute__((ext_vector_type(8)))  float    v8f;
typedef __attribute__((ext_vector_type(4)))  unsigned u32x4;
typedef __attribute__((ext_vector_type(8)))  int      i32x8;
typedef __attribute__((ext_vector_type(4)))  int      i32x4;

#define HIDDEN 2048
#define NB 2
#define SEQ 2048
#define NH 16
#define HD 128
#define ROWS (NB * SEQ) /* 4096 */

// ---------------------------------------------------------------------------
// WMMA fragment loaders (wave32, V_WMMA_F32_16X16X32_F16; contiguous k = 16*(lane/16) + s in both A and B)
__device__ inline v16h load_a_frag(const f16* p, int ld, int lane) {
  const int row = lane & 15;
  const int ko  = (lane >> 4) * 16;
  union { v16h v; h8 h[2]; } u;
  u.h[0] = *(const h8*)(p + row * ld + ko);
  u.h[1] = *(const h8*)(p + row * ld + ko + 8);
  return u.v;
}
__device__ inline v16h load_b_frag(const f16* p, int ld, int lane) {
  const int col = lane & 15;
  const int ko  = (lane >> 4) * 16;
  union { v16h v; h8 h[2]; } u;
  const f16* q = p + col * ld + ko;
  u.h[0] = *(const h8*)(q);
  u.h[1] = *(const h8*)(q + 8);
  return u.v;
}

// ---------------------------------------------------------------------------
// TDM: DMA one dense 2D f16 tile (tile_d0 x tile_d1 elems, row stride `stride`
// elems, row-major) from global memory into LDS at byte offset `lds_off`.
// D# layout per CDNA5 ISA §8.3/§8.4; 6-arg builtin; tracked by TENSORcnt.
__device__ inline void tdm_load_tile_f16(const f16* gptr, unsigned lds_off,
                                         unsigned tile_d0, unsigned tile_d1,
                                         unsigned stride) {
  const unsigned long long va = (unsigned long long)(uintptr_t)gptr;
  u32x4 g0;
  g0[0] = 1u;                                   // count=1, user descriptor
  g0[1] = lds_off;                              // lds_addr (bytes)
  g0[2] = (unsigned)(va & 0xffffffffu);         // global_addr[31:0]
  g0[3] = (unsigned)((va >> 32) & 0x01ffffffu)  // global_addr[56:32]
          | (2u << 30);                         // type = 2 ("image")
  i32x8 g1;
  g1[0] = (int)(1u << 16);                      // data_size = 1 -> 2 bytes
  g1[1] = (int)(0x7fffu << 16);                 // tensor_dim0[15:0] (large)
  g1[2] = (int)(0x7fffu << 16);                 // dim0 hi=0 | tensor_dim1[15:0]
  g1[3] = (int)(tile_d0 << 16);                 // dim1 hi=0 | tile_dim0
  g1[4] = (int)tile_d1;                         // tile_dim1 | tile_dim2=0
  g1[5] = (int)stride;                          // tensor_dim0_stride[31:0]
  g1[6] = 0;                                    // stride hi | dim1_stride lo
  g1[7] = 0;
  const i32x4 z4 = {};
  const i32x8 z8 = {};
  __builtin_amdgcn_tensor_load_to_lds(g0, g1, z4, z4, z8, 0);
}

// ---------------------------------------------------------------------------
__global__ void cast_f32_to_f16(const float* __restrict__ in,
                                f16* __restrict__ out, int n) {
  int i = blockIdx.x * blockDim.x + threadIdx.x;
  int stride = gridDim.x * blockDim.x;
  for (; i < n; i += stride) out[i] = (f16)in[i];
}

// ---------------------------------------------------------------------------
// C[m,n] = sum_k A[m,k] * W[n,k]   (nn.Linear: X @ W^T)
// 4 waves / block, each wave owns a 16x64 tile (4 WMMA accumulators).
template <bool OUT_F32>
__global__ void __launch_bounds__(128, 1)
gemm_xwT(const f16* __restrict__ A, const f16* __restrict__ W,
         void* __restrict__ Cout, int K) {
  const int lane = threadIdx.x & 31;
  const int wid  = threadIdx.x >> 5;
  const int m0 = (blockIdx.x * 4 + wid) * 16;
  const int n0 = blockIdx.y * 64;
  v8f acc0 = {}, acc1 = {}, acc2 = {}, acc3 = {};
  const f16* Ap = A + (size_t)m0 * K;
  for (int k0 = 0; k0 < K; k0 += 32) {
    if (k0 + 32 < K)
      __builtin_prefetch((const void*)(Ap + (lane & 15) * K + k0 + 32), 0, 0);
    v16h a  = load_a_frag(Ap + k0, K, lane);
    v16h b0 = load_b_frag(W + (size_t)(n0 +  0) * K + k0, K, lane);
    v16h b1 = load_b_frag(W + (size_t)(n0 + 16) * K + k0, K, lane);
    v16h b2 = load_b_frag(W + (size_t)(n0 + 32) * K + k0, K, lane);
    v16h b3 = load_b_frag(W + (size_t)(n0 + 48) * K + k0, K, lane);
    acc0 = __builtin_amdgcn_wmma_f32_16x16x32_f16(false, a, false, b0, (short)0, acc0, false, false);
    acc1 = __builtin_amdgcn_wmma_f32_16x16x32_f16(false, a, false, b1, (short)0, acc1, false, false);
    acc2 = __builtin_amdgcn_wmma_f32_16x16x32_f16(false, a, false, b2, (short)0, acc2, false, false);
    acc3 = __builtin_amdgcn_wmma_f32_16x16x32_f16(false, a, false, b3, (short)0, acc3, false, false);
  }
  const int colc = lane & 15;
  const int rb   = (lane >> 4) * 8;
#pragma unroll
  for (int r = 0; r < 8; ++r) {
    const size_t row = (size_t)(m0 + rb + r) * HIDDEN + n0 + colc;
    if constexpr (OUT_F32) {
      float* C = (float*)Cout;
      C[row + 0] = acc0[r]; C[row + 16] = acc1[r];
      C[row + 32] = acc2[r]; C[row + 48] = acc3[r];
    } else {
      f16* C = (f16*)Cout;
      C[row + 0] = (f16)acc0[r]; C[row + 16] = (f16)acc1[r];
      C[row + 32] = (f16)acc2[r]; C[row + 48] = (f16)acc3[r];
    }
  }
}

// ---------------------------------------------------------------------------
// RoPE on Q and K (fp32 math); folds 1/sqrt(HD) attention scale into Q.
__global__ void rope_qk(f16* __restrict__ Q, f16* __restrict__ Kb,
                        const int* __restrict__ pos_ids) {
  int i = blockIdx.x * blockDim.x + threadIdx.x;
  if (i >= NB * SEQ * NH * 64) return;
  const int d = i & 63;
  const int h = (i >> 6) & (NH - 1);
  const int s = (i >> 10) & (SEQ - 1);
  const int b = i >> 21;
  const int row  = b * SEQ + s;
  const int base = row * HIDDEN + h * HD;
  const float pos  = (float)pos_ids[row];
  const float invf = __expf(-(float)d * 0.14391156861f); // 10000^(-2d/128)
  float sn, cs;
  __sincosf(pos * invf, &sn, &cs);
  const float scale = 0.08838834764831845f; // 1/sqrt(128)
  const float q1 = (float)Q[base + d], q2 = (float)Q[base + d + 64];
  Q[base + d]      = (f16)((q1 * cs - q2 * sn) * scale);
  Q[base + d + 64] = (f16)((q2 * cs + q1 * sn) * scale);
  const float k1 = (float)Kb[base + d], k2 = (float)Kb[base + d + 64];
  Kb[base + d]      = (f16)(k1 * cs - k2 * sn);
  Kb[base + d + 64] = (f16)(k2 * cs + k1 * sn);
}

// ---------------------------------------------------------------------------
// Flash-attention: 1 wave per 16-row Q tile; k processed in 32-row steps.
// V tiles are DMA'd global->LDS by the Tensor Data Mover, double-buffered and
// overlapped with the 16 WMMAs + softmax of the current step. All 8 K
// B-fragments are preloaded into distinct registers so the 16 global loads
// clause together instead of serializing load->wait->wmma.
__global__ void __launch_bounds__(128, 1)
attn_kernel(const f16* __restrict__ Q, const f16* __restrict__ Kx,
            const f16* __restrict__ V, f16* __restrict__ O) {
  __shared__ f16   vbuf[4][2][32 * HD]; // 64 KB: per-wave double buffer
  __shared__ float pbuf[4][16 * 32];    //  8 KB
  const int lane = threadIdx.x & 31;
  const int wid  = threadIdx.x >> 5;
  const int qt = blockIdx.x * 4 + wid; // 16-row q tile index, 0..127
  const int h  = blockIdx.y;
  const int b  = blockIdx.z;
  const int colc = lane & 15;
  const int rb   = (lane >> 4) * 8;

  const unsigned voff[2] = {(unsigned)(uintptr_t)&vbuf[wid][0][0],
                            (unsigned)(uintptr_t)&vbuf[wid][1][0]};
  const f16* Vb = V + (size_t)(b * SEQ) * HIDDEN + h * HD;

  const f16* Qp = Q + (size_t)(b * SEQ + qt * 16) * HIDDEN + h * HD;
  v16h aq[4];
#pragma unroll
  for (int c = 0; c < 4; ++c) aq[c] = load_a_frag(Qp + 32 * c, HIDDEN, lane);

  v8f o[8] = {};
  float mrow[8], lrow[8];
#pragma unroll
  for (int r = 0; r < 8; ++r) { mrow[r] = -1e30f; lrow[r] = 0.f; }

  const int steps = (qt * 16 + 47) >> 5; // k up to and including the diagonal
  // TDM: prefetch first V tile (32 rows x 128 cols, row stride HIDDEN)
  tdm_load_tile_f16(Vb, voff[0], HD, 32, HIDDEN);

  for (int s32 = 0; s32 < steps; ++s32) {
    const int kbase = s32 * 32;
    const int cur = s32 & 1;
    if (s32 + 1 < steps) { // kick next tile, then wait for current
      tdm_load_tile_f16(Vb + (size_t)(kbase + 32) * HIDDEN, voff[cur ^ 1],
                        HD, 32, HIDDEN);
      __builtin_amdgcn_s_wait_tensorcnt((short)1);
    } else {
      __builtin_amdgcn_s_wait_tensorcnt((short)0);
    }

    // preload all 8 K B-fragments (distinct registers -> overlapping loads)
    const f16* Kp0 = Kx + (size_t)(b * SEQ + kbase +  0) * HIDDEN + h * HD;
    const f16* Kp1 = Kx + (size_t)(b * SEQ + kbase + 16) * HIDDEN + h * HD;
    v16h bk0[4], bk1[4];
#pragma unroll
    for (int c = 0; c < 4; ++c) bk0[c] = load_b_frag(Kp0 + 32 * c, HIDDEN, lane);
#pragma unroll
    for (int c = 0; c < 4; ++c) bk1[c] = load_b_frag(Kp1 + 32 * c, HIDDEN, lane);

    // scores: two 16x16 subtiles over the 32 k rows (8 WMMAs)
    v8f sc[2] = {};
#pragma unroll
    for (int c = 0; c < 4; ++c)
      sc[0] = __builtin_amdgcn_wmma_f32_16x16x32_f16(false, aq[c], false, bk0[c],
                                                     (short)0, sc[0], false, false);
#pragma unroll
    for (int c = 0; c < 4; ++c)
      sc[1] = __builtin_amdgcn_wmma_f32_16x16x32_f16(false, aq[c], false, bk1[c],
                                                     (short)0, sc[1], false, false);

    // causal mask (additive -inf surrogate), uniform control flow
#pragma unroll
    for (int t = 0; t < 2; ++t) {
      const int kcol = kbase + t * 16 + colc;
#pragma unroll
      for (int r = 0; r < 8; ++r) {
        const int qrow = qt * 16 + rb + r;
        if (kcol > qrow) sc[t][r] = -1e30f;
      }
    }
    // online softmax per q row (16-lane-group reductions), write P to LDS
#pragma unroll
    for (int r = 0; r < 8; ++r) {
      float v = fmaxf(sc[0][r], sc[1][r]);
#pragma unroll
      for (int off = 8; off >= 1; off >>= 1) v = fmaxf(v, __shfl_xor(v, off, 16));
      const float newm = fmaxf(mrow[r], v);
      const float fs = __expf(mrow[r] - newm);
      const float p0 = __expf(sc[0][r] - newm);
      const float p1 = __expf(sc[1][r] - newm);
      float rs = p0 + p1;
#pragma unroll
      for (int off = 8; off >= 1; off >>= 1) rs += __shfl_xor(rs, off, 16);
      lrow[r] = lrow[r] * fs + rs;
      mrow[r] = newm;
#pragma unroll
      for (int dc = 0; dc < 8; ++dc) o[dc][r] *= fs;
      pbuf[wid][(rb + r) * 32 + colc]      = p0;
      pbuf[wid][(rb + r) * 32 + 16 + colc] = p1;
    }
    // P (16x32) as WMMA A-fragment: lane L holds row M=L&15
    v16h ap;
#pragma unroll
    for (int e = 0; e < 16; ++e) {
      const int kk = e + ((lane >> 4) << 4);
      ap[e] = (f16)pbuf[wid][colc * 32 + kk];
    }
    // PV: B fragment of V^T per 16-wide d chunk (transposed read from LDS)
#pragma unroll
    for (int dc = 0; dc < 8; ++dc) {
      v16h bv;
#pragma unroll
      for (int e = 0; e < 16; ++e) {
        const int kk = (lane < 16 ? e : 16 + e);
        bv[e] = vbuf[wid][cur][kk * HD + dc * 16 + colc];
      }
      o[dc] = __builtin_amdgcn_wmma_f32_16x16x32_f16(false, ap, false, bv,
                                                     (short)0, o[dc], false, false);
    }
  }
  // normalize and store attn output in [B, S, H*D] (f16)
  f16* Op = O + (size_t)(b * SEQ + qt * 16) * HIDDEN + h * HD;
#pragma unroll
  for (int r = 0; r < 8; ++r) {
    const float inv = 1.f / lrow[r];
#pragma unroll
    for (int dc = 0; dc < 8; ++dc)
      Op[(rb + r) * HIDDEN + dc * 16 + colc] = (f16)(o[dc][r] * inv);
  }
}

// ---------------------------------------------------------------------------
extern "C" void kernel_launch(void* const* d_in, const int* in_sizes, int n_in,
                              void* d_out, int out_size, void* d_ws, size_t ws_size,
                              hipStream_t stream) {
  (void)in_sizes; (void)n_in; (void)out_size; (void)ws_size;
  const float* hs  = (const float*)d_in[0];
  // d_in[1] = attention_mask (pure causal; reproduced analytically)
  const int*   pid = (const int*)d_in[2];
  const float* Wq  = (const float*)d_in[3];
  const float* Wk  = (const float*)d_in[4];
  const float* Wv  = (const float*)d_in[5];
  const float* Wo  = (const float*)d_in[6];

  char* ws = (char*)d_ws;
  const size_t XEL = (size_t)ROWS * HIDDEN;
  const size_t WEL = (size_t)HIDDEN * HIDDEN;
  const size_t XB = XEL * sizeof(f16); // 16 MiB
  const size_t WB = WEL * sizeof(f16); //  8 MiB
  f16* Xh  = (f16*)(ws);               // X (f16); reused as attn buffer
  f16* Wqh = (f16*)(ws + XB);
  f16* Wkh = (f16*)(ws + XB + 1 * WB);
  f16* Wvh = (f16*)(ws + XB + 2 * WB);
  f16* Woh = (f16*)(ws + XB + 3 * WB);
  f16* Qh  = (f16*)(ws + XB + 4 * WB);
  f16* Kh  = (f16*)(ws + 2 * XB + 4 * WB);
  f16* Vh  = (f16*)(ws + 3 * XB + 4 * WB); // total 96 MiB

  // 1) down-convert inputs to f16
  cast_f32_to_f16<<<dim3(4096), 256, 0, stream>>>(hs, Xh, (int)XEL);
  cast_f32_to_f16<<<dim3(2048), 256, 0, stream>>>(Wq, Wqh, (int)WEL);
  cast_f32_to_f16<<<dim3(2048), 256, 0, stream>>>(Wk, Wkh, (int)WEL);
  cast_f32_to_f16<<<dim3(2048), 256, 0, stream>>>(Wv, Wvh, (int)WEL);
  cast_f32_to_f16<<<dim3(2048), 256, 0, stream>>>(Wo, Woh, (int)WEL);

  // 2) Q/K/V projections (f16 WMMA, f32 accumulate)
  const dim3 ggrid(ROWS / 64, HIDDEN / 64);
  gemm_xwT<false><<<ggrid, 128, 0, stream>>>(Xh, Wqh, Qh, HIDDEN);
  gemm_xwT<false><<<ggrid, 128, 0, stream>>>(Xh, Wkh, Kh, HIDDEN);
  gemm_xwT<false><<<ggrid, 128, 0, stream>>>(Xh, Wvh, Vh, HIDDEN);

  // 3) RoPE (+1/sqrt(HD) folded into Q)
  rope_qk<<<dim3((NB * SEQ * NH * 64) / 256), 256, 0, stream>>>(Qh, Kh, pid);

  // 4) causal flash attention (TDM-fed V tiles) -> attn buffer (reuses Xh)
  attn_kernel<<<dim3(SEQ / 64, NH, NB), 128, 0, stream>>>(Qh, Kh, Vh, Xh);

  // 5) output projection, f32 result straight to d_out
  gemm_xwT<true><<<ggrid, 128, 0, stream>>>(Xh, Woh, d_out, HIDDEN);
}
